// MambaLiteBlock_19550691131979
// MI455X (gfx1250) — matrix operand order repaired and checked by reading, not compiled, not run on hardware
//
#include <hip/hip_runtime.h>

// ---------------------------------------------------------------------------
// MambaLite block for MI455X (gfx1250), wave32 + WMMA bf16 + async LDS fills.
//   B=4, S=4096, DIM=1024, SDIM=256, BS=16384
// ---------------------------------------------------------------------------

#define BDIM   4
#define SDIM_S 4096
#define DDIM   1024
#define NDIM   256
#define BS     (BDIM * SDIM_S)     // 16384 token rows
#define NCAT   2816                // 1024 + 256 + 256 + 256 + 1024
#define CHUNK  256
#define NCHUNK (SDIM_S / CHUNK)    // 16

typedef __attribute__((ext_vector_type(16))) __bf16 v16bf;
typedef __attribute__((ext_vector_type(8)))  float  v8f;
typedef __attribute__((ext_vector_type(4)))  unsigned int u32x4;

union Frag16 { v16bf v; u32x4 u[2]; };

__device__ __forceinline__ float sigf(float x) { return 1.0f / (1.0f + __expf(-x)); }

// CDNA5 async global->LDS copy (VDST = per-lane LDS byte address, ASYNCcnt).
__device__ __forceinline__ void async_ld_b128(const void* gptr, unsigned lds_off) {
    asm volatile("global_load_async_to_lds_b128 %0, %1, off"
                 :: "v"(lds_off), "v"((unsigned long long)gptr) : "memory");
}
__device__ __forceinline__ void wait_async8() {
    asm volatile("s_wait_asynccnt 0x8" ::: "memory");
}
__device__ __forceinline__ void wait_async0() {
    asm volatile("s_wait_asynccnt 0x0" ::: "memory");
}

// ---------------------------------------------------------------------------
// Kernel 1: LayerNorm per token row (block of 256 threads handles D=1024)
// ---------------------------------------------------------------------------
__global__ __launch_bounds__(256)
void ln_kernel(const float* __restrict__ x, const float* __restrict__ gamma,
               const float* __restrict__ beta, float* __restrict__ h) {
    const int row = blockIdx.x;
    const int tid = threadIdx.x;
    const float* xr = x + (size_t)row * DDIM;
    __shared__ float red[256];

    float v[4];
    float s = 0.f;
#pragma unroll
    for (int i = 0; i < 4; ++i) { v[i] = xr[tid + i * 256]; s += v[i]; }
    red[tid] = s; __syncthreads();
    for (int off = 128; off > 0; off >>= 1) {
        if (tid < off) red[tid] += red[tid + off];
        __syncthreads();
    }
    const float mu = red[0] * (1.0f / DDIM);
    __syncthreads();

    float ss = 0.f;
#pragma unroll
    for (int i = 0; i < 4; ++i) { float d = v[i] - mu; ss += d * d; }
    red[tid] = ss; __syncthreads();
    for (int off = 128; off > 0; off >>= 1) {
        if (tid < off) red[tid] += red[tid + off];
        __syncthreads();
    }
    const float rstd = rsqrtf(red[0] * (1.0f / DDIM) + 1e-5f);

    float* hr = h + (size_t)row * DDIM;
#pragma unroll
    for (int i = 0; i < 4; ++i) {
        int c = tid + i * 256;
        hr[c] = (v[i] - mu) * rstd * gamma[c] + beta[c];
    }
}

// ---------------------------------------------------------------------------
// Kernel 2: depthwise conv k=3 pad=1 + bias, fp32 h -> bf16 mixed
// ---------------------------------------------------------------------------
__global__ __launch_bounds__(256)
void conv_kernel(const float* __restrict__ h, const float* __restrict__ mw,
                 const float* __restrict__ mb, __bf16* __restrict__ mixed) {
    const size_t total = (size_t)BS * DDIM;
    for (size_t idx = (size_t)blockIdx.x * blockDim.x + threadIdx.x;
         idx < total; idx += (size_t)gridDim.x * blockDim.x) {
        int c = (int)(idx % DDIM);
        size_t rowflat = idx / DDIM;            // b*S + s
        int s = (int)(rowflat % SDIM_S);
        float hm = (s > 0)          ? h[idx - DDIM] : 0.f;
        float h0 = h[idx];
        float hp = (s < SDIM_S - 1) ? h[idx + DDIM] : 0.f;
        float m  = mw[c * 3 + 0] * hm + mw[c * 3 + 1] * h0 + mw[c * 3 + 2] * hp + mb[c];
        mixed[idx] = (__bf16)m;
    }
}

// ---------------------------------------------------------------------------
// Kernel 3: transpose + fp32->bf16 weight conversion: Wt[n*K+k] = W[k*N+n]
// ---------------------------------------------------------------------------
__global__ __launch_bounds__(256)
void wcvt_kernel(const float* __restrict__ W, __bf16* __restrict__ Wt,
                 int K, int N) {
    const size_t total = (size_t)K * N;
    for (size_t idx = (size_t)blockIdx.x * blockDim.x + threadIdx.x;
         idx < total; idx += (size_t)gridDim.x * blockDim.x) {
        int n = (int)(idx / K);
        int k = (int)(idx % K);
        Wt[idx] = (__bf16)W[(size_t)k * N + n];
    }
}

// ---------------------------------------------------------------------------
// WMMA GEMM: C[M,N] = A[M,K](bf16,row) x Bt[N,K](bf16, N-major).
// Block tile 128x128, 8 waves (4M x 2N), wave tile 32x64 (2x4 WMMA 16x16x32).
// K-step 64 per LDS stage (2 chunks) => one barrier pair per 16 WMMA.
// Double-buffered LDS fed by global_load_async_to_lds_b128 (ASYNCcnt).
// ---------------------------------------------------------------------------
struct EpiP {
    const float *bi, *bsin, *bd, *bsel, *bg;
    __bf16 *u; float *xs; float *dec; float *sel; __bf16 *g;
    const float *bso; const __bf16 *gin, *uin; __bf16 *h2;
    const float *bout, *xres; float *out;
};

#define LDS_STRIDE 72          // 64 bf16 + 8 pad (144B rows, 16B-aligned chunks)
#define TILE_SH (128 * LDS_STRIDE)

template <int MODE>
__global__ __launch_bounds__(256)
void gemm_wmma(const __bf16* __restrict__ A, const __bf16* __restrict__ Bt,
               int M, int N, int K, EpiP p) {
    __shared__ __attribute__((aligned(16))) unsigned short As[2][TILE_SH];
    __shared__ __attribute__((aligned(16))) unsigned short Bs[2][TILE_SH];

    const int tid   = threadIdx.x;
    const int wave  = tid >> 5;
    const int lane  = tid & 31;
    const int waveM = wave & 3;            // 4 waves * 32 = 128 rows
    const int waveN = wave >> 2;           // 2 waves * 64 = 128 cols
    const int mBase = blockIdx.y * 128;
    const int nBase = blockIdx.x * 128;

    v8f zero = {0.f, 0.f, 0.f, 0.f, 0.f, 0.f, 0.f, 0.f};
    v8f acc[2][4];
#pragma unroll
    for (int mt = 0; mt < 2; ++mt)
#pragma unroll
        for (int nt = 0; nt < 4; ++nt) acc[mt][nt] = zero;

    // Cooperative async fill: 2 threads per row, each moving 64B (4 x b128)
    // of A and of B per 64-wide K stage.
    const int ldRow = tid >> 1;            // 0..127
    const int ldCol = (tid & 1) * 32;      // element offset 0 / 32
    const __bf16* gA = A  + (size_t)(mBase + ldRow) * K + ldCol;
    const __bf16* gB = Bt + (size_t)(nBase + ldRow) * K + ldCol;
    const unsigned ldsA0 = (unsigned)(size_t)&As[0][ldRow * LDS_STRIDE + ldCol];
    const unsigned ldsB0 = (unsigned)(size_t)&Bs[0][ldRow * LDS_STRIDE + ldCol];

    auto issue = [&](int buf, int k0) {
        const unsigned la = ldsA0 + (unsigned)buf * (TILE_SH * 2);
        const unsigned lb = ldsB0 + (unsigned)buf * (TILE_SH * 2);
        async_ld_b128(gA + k0,      la);
        async_ld_b128(gA + k0 + 8,  la + 16);
        async_ld_b128(gA + k0 + 16, la + 32);
        async_ld_b128(gA + k0 + 24, la + 48);
        async_ld_b128(gB + k0,      lb);
        async_ld_b128(gB + k0 + 8,  lb + 16);
        async_ld_b128(gB + k0 + 16, lb + 32);
        async_ld_b128(gB + k0 + 24, lb + 48);
    };

    const int kg = lane >> 4;              // K-group select (ISA frag layout)
    const int rA = lane & 15;

    auto compute = [&](int par) {
        const unsigned short* Ab = As[par];
        const unsigned short* Bb = Bs[par];
#pragma unroll
        for (int kc = 0; kc < 2; ++kc) {   // two K=32 chunks per stage
            const int ke = kc * 32;        // element offset within row
            Frag16 bf[4];
#pragma unroll
            for (int nt = 0; nt < 4; ++nt) {
                int col  = waveN * 64 + nt * 16 + rA;
                int base = col * LDS_STRIDE + ke + kg * 8;
                bf[nt].u[0] = *reinterpret_cast<const u32x4*>(&Bb[base]);
                bf[nt].u[1] = *reinterpret_cast<const u32x4*>(&Bb[base + 16]);
            }
#pragma unroll
            for (int mt = 0; mt < 2; ++mt) {
                int rowm = waveM * 32 + mt * 16 + rA;
                Frag16 af;
                af.u[0] = *reinterpret_cast<const u32x4*>(&Ab[rowm * LDS_STRIDE + ke + kg * 8]);
                af.u[1] = *reinterpret_cast<const u32x4*>(&Ab[rowm * LDS_STRIDE + ke + 16 + kg * 8]);
#pragma unroll
                for (int nt = 0; nt < 4; ++nt)
                    acc[mt][nt] = __builtin_amdgcn_wmma_f32_16x16x32_bf16(
                        false, af.v, false, bf[nt].v, (short)0, acc[mt][nt],
                        false, false);
            }
        }
    };

    issue(0, 0);
    int parity = 0;
    int k0 = 0;
    for (; k0 + 64 < K; k0 += 64) {        // steady state: prefetch next stage
        issue(parity ^ 1, k0 + 64);
        wait_async8();                     // previous stage's 8 fills done
        __syncthreads();                   // all waves' fills visible
        compute(parity);
        __syncthreads();                   // reads done before buffer reuse
        parity ^= 1;
    }
    wait_async0();                         // final stage
    __syncthreads();
    compute(parity);

    // Epilogue. C layout: VGPR r holds M = (lane>=16 ? 8 : 0)+r, N = lane&15.
    const int rHalf = (lane >> 4) * 8;
    const int cIn   = lane & 15;
#pragma unroll
    for (int mt = 0; mt < 2; ++mt) {
#pragma unroll
        for (int nt = 0; nt < 4; ++nt) {
#pragma unroll
            for (int vr = 0; vr < 8; ++vr) {
                const int row = mBase + waveM * 32 + mt * 16 + rHalf + vr;
                const int col = nBase + waveN * 64 + nt * 16 + cIn;
                float v = acc[mt][nt][vr];
                if constexpr (MODE == 0) {
                    if (col < 1024) {
                        p.u[(size_t)row * DDIM + col] = (__bf16)tanhf(v + p.bi[col]);
                    } else if (col < 1280) {
                        int c = col - 1024;
                        p.xs[(size_t)row * NDIM + c] = tanhf(v + p.bsin[c]);
                    } else if (col < 1536) {
                        int c = col - 1280;
                        p.dec[(size_t)row * NDIM + c] = sigf(v + p.bd[c]);
                    } else if (col < 1792) {
                        int c = col - 1536;
                        p.sel[(size_t)row * NDIM + c] = sigf(v + p.bsel[c]);
                    } else {
                        int c = col - 1792;
                        p.g[(size_t)row * DDIM + c] = (__bf16)sigf(v + p.bg[c]);
                    }
                } else if constexpr (MODE == 1) {
                    float y  = v + p.bso[col];
                    float gv = (float)p.gin[(size_t)row * DDIM + col];
                    float uv = (float)p.uin[(size_t)row * DDIM + col];
                    p.h2[(size_t)row * DDIM + col] = (__bf16)(gv * y + (1.f - gv) * uv);
                } else {
                    p.out[(size_t)row * DDIM + col] =
                        v + p.bout[col] + p.xres[(size_t)row * DDIM + col];
                }
            }
        }
    }
}

// ---------------------------------------------------------------------------
// Chunked scan (3 passes). Chain length drops 4096 -> 256.
// state = d*state + (1-d)*x   ==>   per chunk: out = P*in + Q
// ---------------------------------------------------------------------------
__global__ __launch_bounds__(256)
void scan_part1(const float* __restrict__ dec, const float* __restrict__ xs,
                float* __restrict__ chP, float* __restrict__ chQ) {
    const int t = blockIdx.x * 256 + threadIdx.x;   // (b*NCHUNK + c)*NDIM + n
    const int n  = t & (NDIM - 1);
    const int bc = t >> 8;
    const int c  = bc & (NCHUNK - 1);
    const int b  = bc >> 4;
    const size_t base = ((size_t)b * SDIM_S + (size_t)c * CHUNK) * NDIM + n;
    float P = 1.f, Q = 0.f;
    for (int s = 0; s < CHUNK; ++s) {
        const size_t off = base + (size_t)s * NDIM;
        float d = dec[off], xv = xs[off];
        P *= d;
        Q = fmaf(d, Q, (1.f - d) * xv);
    }
    chP[t] = P; chQ[t] = Q;
}

__global__ __launch_bounds__(256)
void scan_part2(const float* __restrict__ chP, const float* __restrict__ chQ,
                float* __restrict__ carry) {
    const int t = blockIdx.x * 256 + threadIdx.x;   // b*NDIM + n
    const int n = t & (NDIM - 1);
    const int b = t >> 8;
    float st = 0.f;
    for (int c = 0; c < NCHUNK; ++c) {
        const int idx = (b * NCHUNK + c) * NDIM + n;
        carry[idx] = st;
        st = fmaf(chP[idx], st, chQ[idx]);
    }
}

__global__ __launch_bounds__(256)
void scan_part3(const float* __restrict__ dec, const float* __restrict__ xs,
                const float* __restrict__ sel, const float* __restrict__ carry,
                __bf16* __restrict__ ssel) {
    const int t = blockIdx.x * 256 + threadIdx.x;
    const int n  = t & (NDIM - 1);
    const int bc = t >> 8;
    const int c  = bc & (NCHUNK - 1);
    const int b  = bc >> 4;
    const size_t base = ((size_t)b * SDIM_S + (size_t)c * CHUNK) * NDIM + n;
    float state = carry[t];
    for (int s = 0; s < CHUNK; ++s) {
        const size_t off = base + (size_t)s * NDIM;
        float d = dec[off], xv = xs[off];
        state = fmaf(d, state, (1.f - d) * xv);
        ssel[off] = (__bf16)(sel[off] * state);
    }
}

// ---------------------------------------------------------------------------
// Host launcher
// ---------------------------------------------------------------------------
extern "C" void kernel_launch(void* const* d_in, const int* in_sizes, int n_in,
                              void* d_out, int out_size, void* d_ws, size_t ws_size,
                              hipStream_t stream) {
    const float* x     = (const float*)d_in[0];
    const float* ln_g  = (const float*)d_in[1];
    const float* ln_b  = (const float*)d_in[2];
    const float* mix_w = (const float*)d_in[3];
    const float* mix_b = (const float*)d_in[4];
    const float* Wi    = (const float*)d_in[5];
    const float* bi    = (const float*)d_in[6];
    const float* Wsin  = (const float*)d_in[7];
    const float* bsin  = (const float*)d_in[8];
    const float* Wd    = (const float*)d_in[9];
    const float* bd    = (const float*)d_in[10];
    const float* Wsel  = (const float*)d_in[11];
    const float* bsel  = (const float*)d_in[12];
    const float* Wso   = (const float*)d_in[13];
    const float* bso   = (const float*)d_in[14];
    const float* Wg    = (const float*)d_in[15];
    const float* bg    = (const float*)d_in[16];
    const float* Wout  = (const float*)d_in[17];
    const float* bout  = (const float*)d_in[18];

    char* ws = (char*)d_ws;
    constexpr size_t MB = 1ull << 20;
    float*  h      = (float*)(ws + 0);           // 64MB, dead after conv
    __bf16* u_bf   = (__bf16*)(ws + 0);          // 32MB (reuses h)
    __bf16* g_bf   = (__bf16*)(ws + 32 * MB);    // 32MB (reuses h)
    __bf16* mixed  = (__bf16*)(ws + 64 * MB);    // 32MB, dead after GEMM1
    __bf16* h2_bf  = mixed;                      // reuse
    float*  xs     = (float*)(ws + 96 * MB);     // 16MB
    float*  dec    = (float*)(ws + 112 * MB);    // 16MB
    float*  sel    = (float*)(ws + 128 * MB);    // 16MB
    __bf16* ssel   = (__bf16*)(ws + 144 * MB);   // 8MB
    __bf16* wcat   = (__bf16*)(ws + 152 * MB);   // 5.5MB [2816 x 1024]
    __bf16* wso_t  = (__bf16*)(ws + 158 * MB);   // 0.5MB [1024 x 256]
    __bf16* wout_t = (__bf16*)(ws + 159 * MB);   // 2MB   [1024 x 1024]
    float*  chP    = (float*)(ws + 162 * MB);    // 64KB
    float*  chQ    = (float*)(ws + 163 * MB);    // 64KB
    float*  carry  = (float*)(ws + 164 * MB);    // 64KB

    ln_kernel<<<BS, 256, 0, stream>>>(x, ln_g, ln_b, h);

    auto wgrid = [](size_t n) { return (int)((n + 255) / 256); };
    wcvt_kernel<<<wgrid((size_t)DDIM * DDIM), 256, 0, stream>>>(Wi,   wcat + (size_t)0    * DDIM, DDIM, DDIM);
    wcvt_kernel<<<wgrid((size_t)DDIM * NDIM), 256, 0, stream>>>(Wsin, wcat + (size_t)1024 * DDIM, DDIM, NDIM);
    wcvt_kernel<<<wgrid((size_t)DDIM * NDIM), 256, 0, stream>>>(Wd,   wcat + (size_t)1280 * DDIM, DDIM, NDIM);
    wcvt_kernel<<<wgrid((size_t)DDIM * NDIM), 256, 0, stream>>>(Wsel, wcat + (size_t)1536 * DDIM, DDIM, NDIM);
    wcvt_kernel<<<wgrid((size_t)DDIM * DDIM), 256, 0, stream>>>(Wg,   wcat + (size_t)1792 * DDIM, DDIM, DDIM);
    wcvt_kernel<<<wgrid((size_t)NDIM * DDIM), 256, 0, stream>>>(Wso,  wso_t,  NDIM, DDIM);
    wcvt_kernel<<<wgrid((size_t)DDIM * DDIM), 256, 0, stream>>>(Wout, wout_t, DDIM, DDIM);

    conv_kernel<<<32768, 256, 0, stream>>>(h, mix_w, mix_b, mixed);

    EpiP p0 = {};
    p0.bi = bi; p0.bsin = bsin; p0.bd = bd; p0.bsel = bsel; p0.bg = bg;
    p0.u = u_bf; p0.xs = xs; p0.dec = dec; p0.sel = sel; p0.g = g_bf;
    gemm_wmma<0><<<dim3(NCAT / 128, BS / 128), 256, 0, stream>>>(
        mixed, wcat, BS, NCAT, DDIM, p0);

    scan_part1<<<(BDIM * NCHUNK * NDIM) / 256, 256, 0, stream>>>(dec, xs, chP, chQ);
    scan_part2<<<(BDIM * NDIM) / 256, 256, 0, stream>>>(chP, chQ, carry);
    scan_part3<<<(BDIM * NCHUNK * NDIM) / 256, 256, 0, stream>>>(dec, xs, sel, carry, ssel);

    EpiP p1 = {};
    p1.bso = bso; p1.gin = g_bf; p1.uin = u_bf; p1.h2 = h2_bf;
    gemm_wmma<1><<<dim3(DDIM / 128, BS / 128), 256, 0, stream>>>(
        ssel, wso_t, BS, DDIM, NDIM, p1);

    EpiP p2 = {};
    p2.bout = bout; p2.xres = x; p2.out = (float*)d_out;
    gemm_wmma<2><<<dim3(DDIM / 128, BS / 128), 256, 0, stream>>>(
        h2_bf, wout_t, BS, DDIM, DDIM, p2);
}
